// SupervisedContrastiveLoss_38130719654038
// MI455X (gfx1250) — matrix operand order repaired and checked by reading, not compiled, not run on hardware
//
#include <hip/hip_runtime.h>
#include <hip/hip_bf16.h>
#include <stdint.h>

// ---------------------------------------------------------------------------
// Supervised contrastive loss, B=4096, D=1024, T=1.0
//  sim = E E^T  (bf16 WMMA, f32 accum), fused row reductions, scalar combine.
// ---------------------------------------------------------------------------

#define BATCH 4096
#define DIM   1024
#define NSPLIT 16                       // column splits of the grid
#define ROWS_PER_WG 128                 // 8 waves * 16 rows
#define COLS_PER_SPLIT (BATCH / NSPLIT) // 256
#define COLS_PER_PASS 128               // 8 tiles of 16 per wave pass

typedef __attribute__((ext_vector_type(16))) __bf16 v16bf;
typedef __attribute__((ext_vector_type(8)))  float  v8f;

union FragU { uint4 q[2]; v16bf v; };

__device__ __forceinline__ unsigned short f2bf_rn(float f) {
  unsigned int u = __float_as_uint(f);
  unsigned int r = u + 0x7FFFu + ((u >> 16) & 1u);   // round-to-nearest-even
  return (unsigned short)(r >> 16);
}

// A fragment (16x32 bf16): lane L holds row M=L&15; K chunks at +0 and +16
// starting from K offset (L>>4)*8.  Two contiguous 16B loads.
__device__ __forceinline__ v16bf load_a_frag(const unsigned short* p) {
  FragU u;
  u.q[0] = *(const uint4*)(p);
  u.q[1] = *(const uint4*)(p + 16);
  return u.v;
}

// B fragment (32x16 bf16, B[k][n]=E[col_n][k]): lane L holds col N=L&15;
// K chunks at +0 and +16 starting from K offset (L>>4)*8.  Two 16B loads.
__device__ __forceinline__ v16bf load_b_frag(const unsigned short* p) {
  FragU u;
  u.q[0] = *(const uint4*)(p);
  u.q[1] = *(const uint4*)(p + 16);
  return u.v;
}

// ---------------------------------------------------------------------------
// Kernel 1: f32 -> bf16 conversion of the embedding matrix.
// ---------------------------------------------------------------------------
__global__ __launch_bounds__(256) void convert_bf16_kernel(
    const float* __restrict__ in, unsigned short* __restrict__ out) {
  int idx = (blockIdx.x * 256 + threadIdx.x) * 4;
  float4 f = *(const float4*)(in + idx);
  ushort4 o;
  o.x = f2bf_rn(f.x);
  o.y = f2bf_rn(f.y);
  o.z = f2bf_rn(f.z);
  o.w = f2bf_rn(f.w);
  *(ushort4*)(out + idx) = o;
}

// ---------------------------------------------------------------------------
// Kernel 2: fused GEMM (E E^T) + per-row reductions.
//  grid = (NSPLIT, BATCH/ROWS_PER_WG), block = 256 (8 waves).
//  Wave w owns rows [rblk*128 + w*16, +16), columns [split*256, +256).
//  Writes deterministic per-(split,row) partials: expsum (excl. diag) and
//  possum (sum of sim over cols with label==0, excl. diag).
// ---------------------------------------------------------------------------
__global__ __launch_bounds__(256) void scl_gemm_kernel(
    const unsigned short* __restrict__ Ebf,
    const int* __restrict__ labels,
    float* __restrict__ evp,   // [NSPLIT][BATCH]
    float* __restrict__ pvp) { // [NSPLIT][BATCH]
  const int split = blockIdx.x;
  const int rblk  = blockIdx.y;
  const int wave  = threadIdx.x >> 5;
  const int lane  = threadIdx.x & 31;
  const int m     = lane & 15;     // row within A tile / col within B tile
  const int hi    = lane >> 4;     // half-wave select

  const int rb = rblk * ROWS_PER_WG + wave * 16;

  const unsigned short* abase = Ebf + (size_t)(rb + m) * DIM + hi * 8;
  const unsigned short* bbase = Ebf + (size_t)m * DIM + hi * 8;

  float ev[8], pv[8];
#pragma unroll
  for (int r = 0; r < 8; ++r) { ev[r] = 0.0f; pv[r] = 0.0f; }

  const int cbeg = split * COLS_PER_SPLIT;
  for (int cb = cbeg; cb < cbeg + COLS_PER_SPLIT; cb += COLS_PER_PASS) {
    v8f acc[8];
#pragma unroll
    for (int t = 0; t < 8; ++t) acc[t] = {};

    for (int kc = 0; kc < DIM; kc += 32) {
      v16bf a = load_a_frag(abase + kc);
      v16bf b[8];
#pragma unroll
      for (int t = 0; t < 8; ++t)
        b[t] = load_b_frag(bbase + (size_t)(cb + t * 16) * DIM + kc);
#pragma unroll
      for (int t = 0; t < 8; ++t)
        acc[t] = __builtin_amdgcn_wmma_f32_16x16x32_bf16(
            false, a, false, b[t], (short)0, acc[t], false, false);
    }

    // Fold the eight finished 16x16 tiles into per-row partial sums.
    // C/D layout: lane L, reg r -> M = r + 8*(L>>4), N = L&15.
#pragma unroll
    for (int t = 0; t < 8; ++t) {
      const int ng = cb + t * 16 + m;          // global column of this lane
      const int lp = (labels[ng] == 0);
#pragma unroll
      for (int r = 0; r < 8; ++r) {
        const int mg = rb + r + hi * 8;        // global row of this element
        const float s = acc[t][r];             // sim (temperature == 1)
        if (mg != ng) {
          ev[r] += __expf(s);
          if (lp) pv[r] += s;
        }
      }
    }
  }

  // Reduce over the 16 lanes of each half-wave (rows rb..rb+7 live in lanes
  // 0-15, rows rb+8..rb+15 in lanes 16-31), then one lane per half writes.
#pragma unroll
  for (int r = 0; r < 8; ++r) {
    float e = ev[r], p = pv[r];
#pragma unroll
    for (int off = 8; off >= 1; off >>= 1) {
      e += __shfl_xor(e, off, 32);
      p += __shfl_xor(p, off, 32);
    }
    if (m == 0) {
      const int row = rb + r + hi * 8;
      evp[split * BATCH + row] = e;
      pvp[split * BATCH + row] = p;
    }
  }
}

// ---------------------------------------------------------------------------
// Kernel 3: combine partials -> scalar loss (single workgroup, deterministic).
// ---------------------------------------------------------------------------
__global__ __launch_bounds__(256) void scl_final_kernel(
    const int* __restrict__ labels,
    const float* __restrict__ evp,
    const float* __restrict__ pvp,
    float* __restrict__ out) {
  __shared__ float s_tot[256];
  __shared__ int   s_cnt[256];
  const int t = threadIdx.x;

  int cnt = 0;
  for (int i = t; i < BATCH; i += 256) cnt += (labels[i] == 0);
  s_cnt[t] = cnt;
  __syncthreads();
  for (int off = 128; off >= 1; off >>= 1) {
    if (t < off) s_cnt[t] += s_cnt[t + off];
    __syncthreads();
  }
  const int n_normal = s_cnt[0];
  __syncthreads();

  float tot = 0.0f;
  for (int i = t; i < BATCH; i += 256) {
    float e = 0.0f, p = 0.0f;
#pragma unroll
    for (int s = 0; s < NSPLIT; ++s) {
      e += evp[s * BATCH + i];
      p += pvp[s * BATCH + i];
    }
    const int is_norm = (labels[i] == 0);
    const int c = n_normal - is_norm;          // # positives for row i
    float row_loss = 0.0f;
    if (c > 0) row_loss = logf(e) - p / (float)c;
    if (is_norm) tot += row_loss;
  }
  s_tot[t] = tot;
  __syncthreads();
  for (int off = 128; off >= 1; off >>= 1) {
    if (t < off) s_tot[t] += s_tot[t + off];
    __syncthreads();
  }
  if (t == 0) out[0] = (n_normal > 0) ? s_tot[0] / (float)n_normal : 0.0f;
}

// ---------------------------------------------------------------------------
extern "C" void kernel_launch(void* const* d_in, const int* in_sizes, int n_in,
                              void* d_out, int out_size, void* d_ws, size_t ws_size,
                              hipStream_t stream) {
  const float* emb    = (const float*)d_in[0];
  const int*   labels = (const int*)d_in[1];
  float*       out    = (float*)d_out;

  // Workspace layout: bf16 copy of E, then per-split partial sums.
  unsigned short* Ebf = (unsigned short*)d_ws;
  float* evp = (float*)((char*)d_ws + (size_t)BATCH * DIM * 2);
  float* pvp = evp + NSPLIT * BATCH;

  // 1) f32 -> bf16 (4096x1024 elems, 4 per thread).
  convert_bf16_kernel<<<(BATCH * DIM) / (4 * 256), 256, 0, stream>>>(emb, Ebf);

  // 2) fused WMMA GEMM + row reductions.
  dim3 grid(NSPLIT, BATCH / ROWS_PER_WG);
  scl_gemm_kernel<<<grid, 256, 0, stream>>>(Ebf, labels, evp, pvp);

  // 3) scalar combine.
  scl_final_kernel<<<1, 256, 0, stream>>>(labels, evp, pvp, out);
}
